// MultiHeadAttention_2972117368885
// MI455X (gfx1250) — matrix operand order repaired and checked by reading, not compiled, not run on hardware
//
#include <hip/hip_runtime.h>
#include <hip/hip_bf16.h>

// ---------------------------------------------------------------------------
// MHA forward for MI455X (gfx1250), wave32 + v_wmma_f32_16x16x32_bf16.
// Pipeline:
//   0) wq/wk/wv/dense weights -> bf16 (once; 2MB each)
//   1) Q = (q @ wq^T + bq) * 1/sqrt(64)  -> bf16 [B,H,S,64]
//   2) K = (k @ wk^T + bk)               -> bf16 [B,H,S,64]
//   3) V = (v @ wv^T + bv)               -> bf16 [B,H,64,S]  (transposed)
//   4) fused: logits(WMMA)->softmax->P store (d_out) -> AV(WMMA) -> ctx bf16 [B,S,1024]
//   5) out = ctx @ dense^T + bd          -> f32 d_out
// Workspace: 4 x 16MB activations + 4 x 2MB weights = 72MB.
// ---------------------------------------------------------------------------

typedef __attribute__((ext_vector_type(16))) __bf16 v16bf;
typedef __attribute__((ext_vector_type(8)))  __bf16 v8bf;
typedef __attribute__((ext_vector_type(4)))  __bf16 v4bf;
typedef __attribute__((ext_vector_type(8)))  float  v8f;
typedef __attribute__((ext_vector_type(4)))  float  f32x4;

#define S_LEN 1024
#define DM    1024
#define NHEAD 16
#define DEPTH 64
#define BATCH 8
#define MROWS (BATCH * S_LEN)   // 8192

__device__ __forceinline__ v8f WMMA(v16bf a, v16bf b, v8f c) {
  return __builtin_amdgcn_wmma_f32_16x16x32_bf16(false, a, false, b, (short)0, c,
                                                 false, false);
}

// A fragment (16x32 bf16, MxK). Lane = 16*g + row. Elements 0..7 -> K=g*8+j,
// elements 8..15 -> K=16+g*8+j.  p points at row start + kbase.
__device__ __forceinline__ v16bf fragA_bf16(const __bf16* p, int g) {
  p += g * 8;
  v8bf lo = *(const v8bf*)p;
  v8bf hi = *(const v8bf*)(p + 16);
  v16bf r;
#pragma unroll
  for (int i = 0; i < 8; i++) { r[i] = lo[i]; r[i + 8] = hi[i]; }
  return r;
}

__device__ __forceinline__ v16bf fragA_f32(const float* p, int g) {
  p += g * 8;
  f32x4 u0 = ((const f32x4*)p)[0];
  f32x4 u1 = ((const f32x4*)p)[1];
  f32x4 u2 = ((const f32x4*)(p + 16))[0];
  f32x4 u3 = ((const f32x4*)(p + 16))[1];
  v16bf r;
#pragma unroll
  for (int i = 0; i < 4; i++) {
    r[i]      = (__bf16)u0[i];
    r[i + 4]  = (__bf16)u1[i];
    r[i + 8]  = (__bf16)u2[i];
    r[i + 12] = (__bf16)u3[i];
  }
  return r;
}

// B fragment (32x16 bf16, KxN). Lane = 16*g + col. Elements 0..7 -> K=g*8+j,
// elements 8..15 -> K=16+g*8+j. p points at col-row start + kbase.
__device__ __forceinline__ v16bf fragB_bf16(const __bf16* p, int g) {
  p += g * 8;
  v8bf lo = *(const v8bf*)p;
  v8bf hi = *(const v8bf*)(p + 16);
  v16bf r;
#pragma unroll
  for (int i = 0; i < 8; i++) { r[i] = lo[i]; r[i + 8] = hi[i]; }
  return r;
}

// ---------------------------------------------------------------------------
// Elementwise f32 -> bf16 (4 elements per thread).
// ---------------------------------------------------------------------------
__global__ void __launch_bounds__(256)
cvt_f32_bf16(const float* __restrict__ in, unsigned short* __restrict__ outv, int n) {
  const int i = (blockIdx.x * 256 + threadIdx.x) * 4;
  if (i < n) {
    f32x4 u = *(const f32x4*)(in + i);
    v4bf r;
    r[0] = (__bf16)u[0]; r[1] = (__bf16)u[1];
    r[2] = (__bf16)u[2]; r[3] = (__bf16)u[3];
    *(v4bf*)((__bf16*)outv + i) = r;
  }
}

// ---------------------------------------------------------------------------
// GEMM: out = X[M,K] @ W[N,K]^T + bias, bf16 WMMA, f32 accumulate.
// W is pre-converted bf16 [N,K]; X is f32 (inputs) or bf16 (ctx).
// MODE 0: bf16 out [B,H,S,64] (scale applied after bias; used for Q/K)
// MODE 1: bf16 out [B,H,64,S] (V transposed)
// MODE 2: f32  out [M,N]      (final dense)
// Block: 256 threads = 8 waves in 2x4; each wave owns a 32x32 tile (2x2 WMMA).
// Grid: (M/64, N/128).
// ---------------------------------------------------------------------------
template <bool A_BF16, int MODE>
__global__ void __launch_bounds__(256)
gemm_xwT(const void* __restrict__ Xv, const unsigned short* __restrict__ Wb,
         const float* __restrict__ bias, void* __restrict__ outv, float scale) {
  const int lane  = threadIdx.x & 31;
  const int wave  = threadIdx.x >> 5;
  const int rA    = lane & 15;
  const int g     = lane >> 4;
  const int mBase = blockIdx.x * 64 + (wave >> 2) * 32;
  const int nBase = blockIdx.y * 128 + (wave & 3) * 32;

  v8f acc00 = {}, acc01 = {}, acc10 = {}, acc11 = {};

  const float*  Xf = (const float*)Xv;
  const __bf16* Xb = (const __bf16*)Xv;
  const __bf16* W  = (const __bf16*)Wb;

  for (int k = 0; k < DM; k += 32) {
    v16bf a0, a1;
    if (A_BF16) {
      a0 = fragA_bf16(Xb + (size_t)(mBase + rA) * DM + k, g);
      a1 = fragA_bf16(Xb + (size_t)(mBase + 16 + rA) * DM + k, g);
    } else {
      a0 = fragA_f32(Xf + (size_t)(mBase + rA) * DM + k, g);
      a1 = fragA_f32(Xf + (size_t)(mBase + 16 + rA) * DM + k, g);
    }
    v16bf b0 = fragB_bf16(W + (size_t)(nBase + rA) * DM + k, g);
    v16bf b1 = fragB_bf16(W + (size_t)(nBase + 16 + rA) * DM + k, g);
    acc00 = WMMA(a0, b0, acc00);
    acc01 = WMMA(a0, b1, acc01);
    acc10 = WMMA(a1, b0, acc10);
    acc11 = WMMA(a1, b1, acc11);
  }

  const int mg = g * 8;
#pragma unroll
  for (int st = 0; st < 4; st++) {
    v8f acc = (st == 0) ? acc00 : (st == 1) ? acc01 : (st == 2) ? acc10 : acc11;
    const int mT = mBase + ((st >> 1) ? 16 : 0);
    const int n  = nBase + ((st & 1) ? 16 : 0) + rA;
    const float bs = bias[n];
#pragma unroll
    for (int r = 0; r < 8; r++) {
      const int m = mT + mg + r;
      float val = (acc[r] + bs) * scale;
      if (MODE == 0) {
        const int b = m >> 10, s = m & 1023, h = n >> 6, d = n & 63;
        ((__bf16*)outv)[((((size_t)(b * NHEAD + h)) << 10) + s) * DEPTH + d] =
            (__bf16)val;
      } else if (MODE == 1) {
        const int b = m >> 10, s = m & 1023, h = n >> 6, d = n & 63;
        ((__bf16*)outv)[(((size_t)((b * NHEAD + h) * DEPTH + d)) << 10) + s] =
            (__bf16)val;
      } else {
        ((float*)outv)[(size_t)m * DM + n] = val;
      }
    }
  }
}

// ---------------------------------------------------------------------------
// Fused attention for one (b,h,16-row strip):
//  phase1: logits 16x1024 -> LDS (8 waves x 8 col-tiles, 2 WMMA each)
//  phase2: masked softmax per row (wave shuffle reductions), stream P to d_out
//  phase3: AV 16x64 (waves 0..3, K=1024 in 32-steps, A from LDS, B=V^T)
// Grid: (S/16, B*H), 256 threads. LDS: 64KB.
// ---------------------------------------------------------------------------
__global__ void __launch_bounds__(256)
attn_fused(const unsigned short* __restrict__ Qh_, const unsigned short* __restrict__ Kh_,
           const unsigned short* __restrict__ VhT_, const float* __restrict__ mask,
           float* __restrict__ attnW, unsigned short* __restrict__ ctx_) {
  __shared__ float lds[16 * S_LEN];  // 64 KB

  const __bf16* Qh  = (const __bf16*)Qh_;
  const __bf16* Kh  = (const __bf16*)Kh_;
  const __bf16* VhT = (const __bf16*)VhT_;
  __bf16*       ctx = (__bf16*)ctx_;

  const int lane  = threadIdx.x & 31;
  const int wave  = threadIdx.x >> 5;
  const int rA    = lane & 15;
  const int g     = lane >> 4;
  const int strip = blockIdx.x;      // 0..63
  const int bh    = blockIdx.y;      // 0..127
  const int b     = bh >> 4;
  const int h     = bh & 15;

  const __bf16* qbase = Qh + ((size_t)bh << 16);   // bh * S*DEPTH
  const __bf16* kbase = Kh + ((size_t)bh << 16);

  // --- phase 1: logits strip into LDS ---
  const v16bf aq0 = fragA_bf16(qbase + (size_t)(strip * 16 + rA) * DEPTH, g);
  const v16bf aq1 = fragA_bf16(qbase + (size_t)(strip * 16 + rA) * DEPTH + 32, g);
#pragma unroll
  for (int t = 0; t < 8; t++) {
    const int col = wave * 128 + t * 16;
    v16bf bk0 = fragB_bf16(kbase + (size_t)(col + rA) * DEPTH, g);
    v16bf bk1 = fragB_bf16(kbase + (size_t)(col + rA) * DEPTH + 32, g);
    v8f acc = {};
    acc = WMMA(aq0, bk0, acc);
    acc = WMMA(aq1, bk1, acc);
#pragma unroll
    for (int r = 0; r < 8; r++) lds[(g * 8 + r) * S_LEN + col + rA] = acc[r];
  }
  __syncthreads();

  // --- phase 2: masked softmax, write P to global ---
  const float* mrow = mask + (size_t)b * S_LEN;
#pragma unroll
  for (int rr = wave * 2; rr < wave * 2 + 2; rr++) {
    float* row = lds + rr * S_LEN;
    float mx = -3.4e38f;
#pragma unroll
    for (int c0 = 0; c0 < S_LEN; c0 += 32) {
      const int c = c0 + lane;
      const float x = row[c] + mrow[c] * (-1e9f);
      row[c] = x;
      mx = fmaxf(mx, x);
    }
#pragma unroll
    for (int off = 16; off > 0; off >>= 1) mx = fmaxf(mx, __shfl_xor(mx, off, 32));
    float sum = 0.f;
#pragma unroll
    for (int c0 = 0; c0 < S_LEN; c0 += 32) {
      const int c = c0 + lane;
      const float e = __expf(row[c] - mx);
      row[c] = e;
      sum += e;
    }
#pragma unroll
    for (int off = 16; off > 0; off >>= 1) sum += __shfl_xor(sum, off, 32);
    const float inv = 1.0f / sum;
    float* gdst = attnW + (((size_t)bh << 10) + strip * 16 + rr) * S_LEN;
#pragma unroll
    for (int c0 = 0; c0 < S_LEN; c0 += 32) {
      const int c = c0 + lane;
      const float p = row[c] * inv;
      row[c] = p;
      gdst[c] = p;
    }
  }
  __syncthreads();

  // --- phase 3: AV (context strip 16x64) ---
  if (wave < 4) {
    const __bf16* vbase = VhT + ((size_t)bh << 16);  // bh * DEPTH*S
    v8f acc = {};
    for (int k = 0; k < S_LEN; k += 32) {
      const v16bf ap = fragA_f32(lds + rA * S_LEN + k, g);  // P from LDS, cvt
      const v16bf bv = fragB_bf16(vbase + (size_t)(wave * 16 + rA) * S_LEN + k, g);
      acc = WMMA(ap, bv, acc);
    }
    const int d = wave * 16 + rA;
#pragma unroll
    for (int r = 0; r < 8; r++) {
      const int s = strip * 16 + g * 8 + r;
      ctx[(((size_t)b << 10) + s) * DM + h * DEPTH + d] = (__bf16)acc[r];
    }
  }
}

// ---------------------------------------------------------------------------
extern "C" void kernel_launch(void* const* d_in, const int* in_sizes, int n_in,
                              void* d_out, int out_size, void* d_ws, size_t ws_size,
                              hipStream_t stream) {
  const float* v    = (const float*)d_in[0];
  const float* k    = (const float*)d_in[1];
  const float* q    = (const float*)d_in[2];
  const float* mask = (const float*)d_in[3];
  const float* wq_w = (const float*)d_in[4];
  const float* wq_b = (const float*)d_in[5];
  const float* wk_w = (const float*)d_in[6];
  const float* wk_b = (const float*)d_in[7];
  const float* wv_w = (const float*)d_in[8];
  const float* wv_b = (const float*)d_in[9];
  const float* dw   = (const float*)d_in[10];
  const float* db   = (const float*)d_in[11];

  float* out   = (float*)d_out;
  float* attnW = out + (size_t)BATCH * S_LEN * DM;  // second tuple element

  char* ws = (char*)d_ws;                           // needs 72 MB
  unsigned short* Qh  = (unsigned short*)(ws);
  unsigned short* Kh  = (unsigned short*)(ws + ((size_t)16 << 20));
  unsigned short* VhT = (unsigned short*)(ws + ((size_t)32 << 20));
  unsigned short* ctx = (unsigned short*)(ws + ((size_t)48 << 20));
  unsigned short* Wq  = (unsigned short*)(ws + ((size_t)64 << 20));
  unsigned short* Wk  = (unsigned short*)(ws + ((size_t)66 << 20));
  unsigned short* Wv  = (unsigned short*)(ws + ((size_t)68 << 20));
  unsigned short* Wd  = (unsigned short*)(ws + ((size_t)70 << 20));

  const dim3 blk(256);
  const int  nW = DM * DM;                          // 1M elements per weight
  const dim3 gcvt((nW / 4 + 255) / 256);            // 1024 blocks

  // Weight conversion (once; stays resident in L2 across GEMM blocks).
  cvt_f32_bf16<<<gcvt, blk, 0, stream>>>(wq_w, Wq, nW);
  cvt_f32_bf16<<<gcvt, blk, 0, stream>>>(wk_w, Wk, nW);
  cvt_f32_bf16<<<gcvt, blk, 0, stream>>>(wv_w, Wv, nW);
  cvt_f32_bf16<<<gcvt, blk, 0, stream>>>(dw,   Wd, nW);

  const dim3 gproj(MROWS / 64, DM / 128);           // (128, 8)

  // Q projection (scale 1/sqrt(depth) folded in), K, V projections.
  gemm_xwT<false, 0><<<gproj, blk, 0, stream>>>((const void*)q, Wq, wq_b,
                                                (void*)Qh, 0.125f);
  gemm_xwT<false, 0><<<gproj, blk, 0, stream>>>((const void*)k, Wk, wk_b,
                                                (void*)Kh, 1.0f);
  gemm_xwT<false, 1><<<gproj, blk, 0, stream>>>((const void*)v, Wv, wv_b,
                                                (void*)VhT, 1.0f);

  // Fused logits -> softmax -> P store -> AV.
  const dim3 gattn(S_LEN / 16, BATCH * NHEAD);      // (64, 128)
  attn_fused<<<gattn, blk, 0, stream>>>(Qh, Kh, VhT, mask, attnW, ctx);

  // Final dense projection to f32 output.
  gemm_xwT<true, 2><<<gproj, blk, 0, stream>>>((const void*)ctx, Wd, db,
                                               (void*)out, 1.0f);
}
